// BiTransformerEncoder_54219667145016
// MI455X (gfx1250) — matrix operand order repaired and checked by reading, not compiled, not run on hardware
//
#include <hip/hip_runtime.h>

#define BATCH   32
#define T_SEQ   512
#define D_MODEL 256
#define NHEADS  8
#define DKH     32
#define DFF     1024

typedef __attribute__((ext_vector_type(16))) __bf16 v16bf;
typedef __attribute__((ext_vector_type(8)))  float  v8f;

// ---------- fp32 -> bf16 (round to nearest even) ----------
__device__ __forceinline__ __bf16 f2bf(float f) {
  union { float f; unsigned u; } un; un.f = f;
  unsigned r = (un.u + 0x7FFFu + ((un.u >> 16) & 1u)) >> 16;
  unsigned short s = (unsigned short)r;
  __bf16 b; __builtin_memcpy(&b, &s, 2);
  return b;
}

__device__ __forceinline__ v8f wmma_bf16(v16bf a, v16bf b, v8f c) {
  return __builtin_amdgcn_wmma_f32_16x16x32_bf16(false, a, false, b, (short)0, c,
                                                 false, false);
}

// ---------- A fragment: 16x32 (MxK) bf16 from row-major bf16 (LDS or global) ----
// lane&15 = M; halves 0..15 -> K = h + 16*hi
__device__ __forceinline__ v16bf a_frag_bf16(const __bf16* base, int ld) {
  int lane = threadIdx.x & 31;
  const __bf16* row = base + (lane & 15) * ld;
  int koff = (lane & 16) ? 16 : 0;
  v16bf a;
#pragma unroll
  for (int h = 0; h < 8; ++h) a[h] = row[h + koff];
#pragma unroll
  for (int h = 0; h < 8; ++h) a[8 + h] = row[8 + h + koff];
  return a;
}

// ---------- B fragment from pre-packed fragment-ordered weights ----------
// Packed layout: tile (tn, tk) at ((tn*ntk + tk)*32 + lane)*16, 16 contiguous bf16
__device__ __forceinline__ v16bf b_frag_pre(const __bf16* wt, int ntk, int tk, int tn) {
  int lane = threadIdx.x & 31;
  const __bf16* p = wt + (((size_t)tn * ntk + tk) * 32 + lane) * 16;
  v16bf b;
  __builtin_memcpy(&b, p, 32);
  return b;
}

// B fragment: 16 contiguous bf16 at a per-lane address (K^T / V^T access)
__device__ __forceinline__ v16bf b_frag_contig(const __bf16* p) {
  v16bf b;
  __builtin_memcpy(&b, p, 32);
  return b;
}

// =====================================================================
// Weight repack: fp32 [K][N] row-major -> bf16 fragment-tiled layout.
// =====================================================================
__global__ void __launch_bounds__(256) pack_weights_kernel(
    const float* __restrict__ src, __bf16* __restrict__ dst, int K, int N)
{
  size_t moff = (size_t)blockIdx.y * K * N;
  src += moff; dst += moff;
  int ntk = K >> 5;
  int chunk = blockIdx.x * 256 + threadIdx.x;   // K*N/16 chunks
  if (chunk >= (K * N) >> 4) return;
  int lane = chunk & 31;
  int tile = chunk >> 5;
  int tk = tile % ntk;
  int tn = tile / ntk;
  int koff = (lane & 16) ? 16 : 0;
  const float* s = src + (size_t)(tk * 32 + koff) * N + tn * 16 + (lane & 15);
  __bf16* d = dst + (size_t)chunk * 16;
#pragma unroll
  for (int h = 0; h < 16; ++h) d[h] = f2bf(s[(size_t)h * N]);
}

// =====================================================================
// QKV projection, dense over 5 behaviors + per-row branchless select.
// Outputs: q,k bf16 [B,H,T,32]; v bf16 TRANSPOSED [B,H,32,T].
// =====================================================================
__global__ void __launch_bounds__(256) proj_kernel(
    const float* __restrict__ x,       // [B,T,256]
    const __bf16* __restrict__ wlin,   // packed [3,5] matrices of 256x256
    const int*   __restrict__ bseq,
    __bf16* __restrict__ outq, __bf16* __restrict__ outk, __bf16* __restrict__ outv)
{
  __shared__ __bf16 xt[16][D_MODEL + 8];
  __shared__ int bsel[16];
  int blk = blockIdx.x;
  int b = blk >> 5, rt = blk & 31;
  int proj = blockIdx.y;
  int tid = threadIdx.x;

  const float* xrow0 = x + (size_t)(b * T_SEQ + rt * 16) * D_MODEL;
  for (int idx = tid * 4; idx < 16 * D_MODEL; idx += 256 * 4) {
    int r = idx >> 8, c = idx & 255;
    float4 f = *(const float4*)(xrow0 + r * D_MODEL + c);
    __bf16* d = &xt[r][c];
    d[0] = f2bf(f.x); d[1] = f2bf(f.y); d[2] = f2bf(f.z); d[3] = f2bf(f.w);
  }
  if (tid < 16) bsel[tid] = bseq[b * T_SEQ + rt * 16 + tid];
  __syncthreads();

  const size_t MSZ = (size_t)D_MODEL * 256;
  const __bf16* W = wlin + (size_t)proj * 5 * MSZ;

  int wave = tid >> 5;
#pragma unroll 1
  for (int cti = 0; cti < 2; ++cti) {
    int ct = wave * 2 + cti;
    v8f acc0 = {}, acc1 = {}, acc2 = {}, acc3 = {}, acc4 = {};
#pragma unroll 1
    for (int tk = 0; tk < 8; ++tk) {
      v16bf afr = a_frag_bf16(&xt[0][tk * 32], D_MODEL + 8);
      __builtin_prefetch(W + (((size_t)ct * 8 + tk + 1) * 32) * 16, 0, 0);
      acc0 = wmma_bf16(afr, b_frag_pre(W + 0 * MSZ, 8, tk, ct), acc0);
      acc1 = wmma_bf16(afr, b_frag_pre(W + 1 * MSZ, 8, tk, ct), acc1);
      acc2 = wmma_bf16(afr, b_frag_pre(W + 2 * MSZ, 8, tk, ct), acc2);
      acc3 = wmma_bf16(afr, b_frag_pre(W + 3 * MSZ, 8, tk, ct), acc3);
      acc4 = wmma_bf16(afr, b_frag_pre(W + 4 * MSZ, 8, tk, ct), acc4);
    }
    int lane = tid & 31;
    int n = lane & 15, hi = (lane & 16) ? 1 : 0;
    int col = ct * 16 + n;
    int hh = col >> 5, dk = col & 31;
    size_t bh = (size_t)b * NHEADS + hh;
#pragma unroll
    for (int i = 0; i < 8; ++i) {
      int row = i + hi * 8;
      int e = bsel[row];
      float v = (e == 0) ? acc0[i] : (e == 1) ? acc1[i]
              : (e == 2) ? acc2[i] : (e == 3) ? acc3[i] : acc4[i];
      int tok = rt * 16 + row;
      __bf16 bv = f2bf(v);
      if (proj == 0)       outq[(bh * T_SEQ + tok) * DKH + dk] = bv;   // uniform branch
      else if (proj == 1)  outk[(bh * T_SEQ + tok) * DKH + dk] = bv;
      else                 outv[(bh * DKH + dk) * T_SEQ + tok] = bv;   // transposed
    }
  }
}

// =====================================================================
// Attention: one wave per (b, h, 16-query tile). Two-pass softmax with
// unnormalized bf16 P in LDS; O rows rescaled by 1/rowsum after P@V.
// =====================================================================
__global__ void __launch_bounds__(32) attn_kernel(
    const __bf16* __restrict__ q,    // [B,H,T,32]
    const __bf16* __restrict__ k,    // [B,H,T,32]
    const __bf16* __restrict__ vt,   // [B,H,32,T]  (transposed)
    const unsigned char* __restrict__ mask,  // [B,T]
    float* __restrict__ out)         // [B,T,256] head-concat
{
  __shared__ float  sc[16][T_SEQ + 4];
  __shared__ __bf16 pb[16][T_SEQ + 8];
  __shared__ float  invr[16];
  int blk = blockIdx.x;
  int qt = blk & 31;
  int bh = blk >> 5;
  int h = bh & 7, b = bh >> 3;
  int lane = threadIdx.x & 31;
  int n = lane & 15, hi = (lane & 16) ? 1 : 0;
  int koffB = hi ? 16 : 0;

  const __bf16* qb = q + ((size_t)bh * T_SEQ + qt * 16) * DKH;
  const __bf16* kb = k + (size_t)bh * T_SEQ * DKH;
  const __bf16* vb = vt + (size_t)bh * DKH * T_SEQ;

  v16bf aq = a_frag_bf16(qb, DKH);
  const float scale = 0.17677669529663687f;  // 1/sqrt(32)

  bool mq[8];
#pragma unroll
  for (int i = 0; i < 8; ++i)
    mq[i] = mask[b * T_SEQ + qt * 16 + i + hi * 8] != 0;

  // scores = (Q K^T) * scale, masked
#pragma unroll 1
  for (int j0 = 0; j0 < T_SEQ; j0 += 16) {
    v16bf bk = b_frag_contig(kb + (size_t)(j0 + n) * DKH + koffB);
    v8f s = {};
    s = wmma_bf16(aq, bk, s);
    int col = j0 + n;
    bool mk = mask[b * T_SEQ + col] != 0;
#pragma unroll
    for (int i = 0; i < 8; ++i) {
      float val = (mq[i] && mk) ? s[i] * scale : -1e30f;
      sc[i + hi * 8][col] = val;
    }
  }
  __syncthreads();

  // softmax (2 passes): lane pair (n, n|16) covers one row's halves
  {
    int row = n;
    float* r = &sc[row][hi * 256];
    float mx = -3.4e38f;
    for (int c = 0; c < 256; ++c) mx = fmaxf(mx, r[c]);
    mx = fmaxf(mx, __shfl_xor(mx, 16));
    float sum = 0.f;
    __bf16* w = &pb[row][hi * 256];
    for (int c = 0; c < 256; ++c) {
      float ev = __expf(r[c] - mx);
      sum += ev;
      w[c] = f2bf(ev);             // unnormalized P
    }
    sum += __shfl_xor(sum, 16);
    if (hi == 0) invr[row] = 1.f / sum;
  }
  __syncthreads();

  // O = (P V) * invr[row]
#pragma unroll 1
  for (int ct = 0; ct < 2; ++ct) {
    v8f acc = {};
    const __bf16* vcol = vb + (size_t)(ct * 16 + n) * T_SEQ + koffB;
#pragma unroll 1
    for (int j0 = 0; j0 < T_SEQ; j0 += 32) {
      v16bf ap = a_frag_bf16(&pb[0][j0], T_SEQ + 8);
      v16bf bv = b_frag_contig(vcol + j0);
      acc = wmma_bf16(ap, bv, acc);
    }
    int col = h * 32 + ct * 16 + n;
#pragma unroll
    for (int i = 0; i < 8; ++i) {
      int row = i + hi * 8;
      int tok = qt * 16 + row;
      out[((size_t)b * T_SEQ + tok) * D_MODEL + col] = acc[i] * invr[row];
    }
  }
}

// =====================================================================
// out = LayerNorm(x + r) * g + beta   (one block per token, 256 threads)
// =====================================================================
__global__ void __launch_bounds__(256) add_ln_kernel(
    const float* __restrict__ x, const float* __restrict__ r,
    const float* __restrict__ g, const float* __restrict__ bet,
    float* __restrict__ out)
{
  __shared__ float red[256];
  int row = blockIdx.x, t = threadIdx.x;
  size_t idx = (size_t)row * D_MODEL + t;
  float v = x[idx] + r[idx];
  red[t] = v;
  __syncthreads();
  for (int s = 128; s > 0; s >>= 1) {
    if (t < s) red[t] += red[t + s];
    __syncthreads();
  }
  float mean = red[0] * (1.0f / D_MODEL);
  __syncthreads();
  float d = v - mean;
  red[t] = d * d;
  __syncthreads();
  for (int s = 128; s > 0; s >>= 1) {
    if (t < s) red[t] += red[t + s];
    __syncthreads();
  }
  float var = red[0] * (1.0f / D_MODEL);
  out[idx] = d * rsqrtf(var + 1e-5f) * g[t] + bet[t];
}

// =====================================================================
// FFN part 1: hid = relu(x @ w1_e + b1_e), branchless per-row select
// =====================================================================
__global__ void __launch_bounds__(256) ffn1_kernel(
    const float* __restrict__ x,      // [B,T,256]
    const __bf16* __restrict__ w1,    // packed 4 matrices 256x1024
    const float* __restrict__ b1,     // [4,1024]
    const int*   __restrict__ bseq,
    __bf16* __restrict__ hid)         // [B,T,1024]
{
  __shared__ __bf16 xt[16][D_MODEL + 8];
  __shared__ int bsel[16];
  int blk = blockIdx.x;
  int b = blk >> 5, rt = blk & 31;
  int tid = threadIdx.x;

  const float* xrow0 = x + (size_t)(b * T_SEQ + rt * 16) * D_MODEL;
  for (int idx = tid * 4; idx < 16 * D_MODEL; idx += 256 * 4) {
    int r = idx >> 8, c = idx & 255;
    float4 f = *(const float4*)(xrow0 + r * D_MODEL + c);
    __bf16* d = &xt[r][c];
    d[0] = f2bf(f.x); d[1] = f2bf(f.y); d[2] = f2bf(f.z); d[3] = f2bf(f.w);
  }
  if (tid < 16) bsel[tid] = bseq[b * T_SEQ + rt * 16 + tid];
  __syncthreads();

  const size_t MSZ = (size_t)D_MODEL * DFF;
  int wave = tid >> 5;
#pragma unroll 1
  for (int cti = 0; cti < 8; ++cti) {
    int ct = wave * 8 + cti;  // 0..63
    v8f acc0 = {}, acc1 = {}, acc2 = {}, acc3 = {};
#pragma unroll 1
    for (int tk = 0; tk < 8; ++tk) {
      v16bf afr = a_frag_bf16(&xt[0][tk * 32], D_MODEL + 8);
      __builtin_prefetch(w1 + (((size_t)ct * 8 + tk + 1) * 32) * 16, 0, 0);
      acc0 = wmma_bf16(afr, b_frag_pre(w1 + 0 * MSZ, 8, tk, ct), acc0);
      acc1 = wmma_bf16(afr, b_frag_pre(w1 + 1 * MSZ, 8, tk, ct), acc1);
      acc2 = wmma_bf16(afr, b_frag_pre(w1 + 2 * MSZ, 8, tk, ct), acc2);
      acc3 = wmma_bf16(afr, b_frag_pre(w1 + 3 * MSZ, 8, tk, ct), acc3);
    }
    int lane = tid & 31;
    int n = lane & 15, hi = (lane & 16) ? 1 : 0;
    int col = ct * 16 + n;
    float bb0 = b1[0 * DFF + col];
    float bb1 = b1[1 * DFF + col];
    float bb2 = b1[2 * DFF + col];
    float bb3 = b1[3 * DFF + col];
#pragma unroll
    for (int i = 0; i < 8; ++i) {
      int row = i + hi * 8;
      int e = bsel[row];
      float t  = (e <= 1) ? acc0[i] : (e == 2) ? acc1[i]
               : (e == 3) ? acc2[i] : acc3[i];
      float bb = (e <= 1) ? bb0 : (e == 2) ? bb1 : (e == 3) ? bb2 : bb3;
      float val = t + bb;
      val = val > 0.f ? val : 0.f;
      val = (e == 0) ? 0.f : val;
      hid[((size_t)(b * T_SEQ) + rt * 16 + row) * DFF + col] = f2bf(val);
    }
  }
}

// =====================================================================
// FFN part 2: p = hid @ w2_e + b2_e, branchless per-row select
// =====================================================================
__global__ void __launch_bounds__(256) ffn2_kernel(
    const __bf16* __restrict__ hid,   // [B,T,1024]
    const __bf16* __restrict__ w2,    // packed 4 matrices 1024x256
    const float* __restrict__ b2,     // [4,256]
    const int*   __restrict__ bseq,
    float* __restrict__ p)            // [B,T,256]
{
  __shared__ __bf16 ht[16][DFF + 8];
  __shared__ int bsel[16];
  int blk = blockIdx.x;
  int b = blk >> 5, rt = blk & 31;
  int tid = threadIdx.x;

  const __bf16* hrow0 = hid + (size_t)(b * T_SEQ + rt * 16) * DFF;
  for (int idx = tid * 8; idx < 16 * DFF; idx += 256 * 8) {
    int r = idx >> 10, c = idx & 1023;
    *(uint4*)&ht[r][c] = *(const uint4*)(hrow0 + r * DFF + c);
  }
  if (tid < 16) bsel[tid] = bseq[b * T_SEQ + rt * 16 + tid];
  __syncthreads();

  const size_t MSZ = (size_t)DFF * D_MODEL;
  int wave = tid >> 5;
#pragma unroll 1
  for (int cti = 0; cti < 2; ++cti) {
    int ct = wave * 2 + cti;  // 0..15
    v8f acc0 = {}, acc1 = {}, acc2 = {}, acc3 = {};
#pragma unroll 1
    for (int tk = 0; tk < 32; ++tk) {
      v16bf afr = a_frag_bf16(&ht[0][tk * 32], DFF + 8);
      __builtin_prefetch(w2 + (((size_t)ct * 32 + tk + 1) * 32) * 16, 0, 0);
      acc0 = wmma_bf16(afr, b_frag_pre(w2 + 0 * MSZ, 32, tk, ct), acc0);
      acc1 = wmma_bf16(afr, b_frag_pre(w2 + 1 * MSZ, 32, tk, ct), acc1);
      acc2 = wmma_bf16(afr, b_frag_pre(w2 + 2 * MSZ, 32, tk, ct), acc2);
      acc3 = wmma_bf16(afr, b_frag_pre(w2 + 3 * MSZ, 32, tk, ct), acc3);
    }
    int lane = tid & 31;
    int n = lane & 15, hi = (lane & 16) ? 1 : 0;
    int col = ct * 16 + n;
    float bb0 = b2[0 * D_MODEL + col];
    float bb1 = b2[1 * D_MODEL + col];
    float bb2 = b2[2 * D_MODEL + col];
    float bb3 = b2[3 * D_MODEL + col];
#pragma unroll
    for (int i = 0; i < 8; ++i) {
      int row = i + hi * 8;
      int e = bsel[row];
      float t  = (e <= 1) ? acc0[i] : (e == 2) ? acc1[i]
               : (e == 3) ? acc2[i] : acc3[i];
      float bb = (e <= 1) ? bb0 : (e == 2) ? bb1 : (e == 3) ? bb2 : bb3;
      float val = t + bb;
      val = (e == 0) ? 0.f : val;
      p[((size_t)(b * T_SEQ) + rt * 16 + row) * D_MODEL + col] = val;
    }
  }
}

// =====================================================================
extern "C" void kernel_launch(void* const* d_in, const int* in_sizes, int n_in,
                              void* d_out, int out_size, void* d_ws, size_t ws_size,
                              hipStream_t stream) {
  const float* h1_in = (const float*)d_in[0];
  const float* h2_in = (const float*)d_in[1];
  const unsigned char* mask = (const unsigned char*)d_in[2];
  const int* bseq = (const int*)d_in[3];
  float* out = (float*)d_out;

  const size_t BTD = (size_t)BATCH * T_SEQ * D_MODEL;  // 4,194,304
  // ---- fp32 scratch ----
  float* ws = (float*)d_ws;
  float* a1 = ws;        float* a2 = a1 + BTD;
  float* x1 = a2 + BTD;  float* x2 = x1 + BTD;
  float* p1 = x2 + BTD;  float* p2 = p1 + BTD;
  float* h1b = p2 + BTD; float* h2b = h1b + BTD;
  // ---- bf16 scratch ----
  __bf16* bws = (__bf16*)(h2b + BTD);
  __bf16* q1 = bws;       __bf16* k1 = q1 + BTD;  __bf16* v1 = k1 + BTD;
  __bf16* q2 = v1 + BTD;  __bf16* k2 = q2 + BTD;  __bf16* v2 = k2 + BTD;
  __bf16* hid = v2 + BTD;                       // B*T*1024
  __bf16* wpk = hid + (size_t)BATCH * T_SEQ * DFF;
  const size_t LIN_SZ = (size_t)15 * D_MODEL * 256;   // 983,040
  const size_t FFW_SZ = (size_t)4 * D_MODEL * DFF;    // 1,048,576
  __bf16* wlin_i[2], *wlin_t[2], *w1i[2], *w2i[2], *w1t[2], *w2t[2];
  {
    __bf16* c = wpk;
    for (int L = 0; L < 2; ++L) {
      wlin_i[L] = c; c += LIN_SZ;
      wlin_t[L] = c; c += LIN_SZ;
      w1i[L] = c; c += FFW_SZ;
      w2i[L] = c; c += FFW_SZ;
      w1t[L] = c; c += FFW_SZ;
      w2t[L] = c; c += FFW_SZ;
    }
  }

  // ---- pack all weights to fragment-ordered bf16 ----
  for (int L = 0; L < 2; ++L) {
    const float* const* pp = (const float* const*)(d_in + 4 + 12 * L);
    pack_weights_kernel<<<dim3(16, 15), 256, 0, stream>>>(pp[0], wlin_i[L], D_MODEL, 256);
    pack_weights_kernel<<<dim3(16, 15), 256, 0, stream>>>(pp[1], wlin_t[L], D_MODEL, 256);
    pack_weights_kernel<<<dim3(64, 4), 256, 0, stream>>>(pp[2], w1i[L], D_MODEL, DFF);
    pack_weights_kernel<<<dim3(64, 4), 256, 0, stream>>>(pp[4], w2i[L], DFF, D_MODEL);
    pack_weights_kernel<<<dim3(64, 4), 256, 0, stream>>>(pp[6], w1t[L], D_MODEL, DFF);
    pack_weights_kernel<<<dim3(64, 4), 256, 0, stream>>>(pp[8], w2t[L], DFF, D_MODEL);
  }

  const float* cur1 = h1_in;
  const float* cur2 = h2_in;

  for (int L = 0; L < 2; ++L) {
    const float* const* pp = (const float* const*)(d_in + 4 + 12 * L);
    const float* i_b1 = pp[3]; const float* i_b2 = pp[5];
    const float* t_b1 = pp[7]; const float* t_b2 = pp[9];
    const float* ng = pp[10];  const float* nb = pp[11];

    dim3 pg(BATCH * 32, 3);
    proj_kernel<<<pg, 256, 0, stream>>>(cur1, wlin_i[L], bseq, q1, k1, v1);
    proj_kernel<<<pg, 256, 0, stream>>>(cur2, wlin_t[L], bseq, q2, k2, v2);

    int ag = BATCH * NHEADS * 32;
    attn_kernel<<<ag, 32, 0, stream>>>(q2, k1, v1, mask, a1);  // attn(tq, ik, iv)
    attn_kernel<<<ag, 32, 0, stream>>>(q1, k2, v2, mask, a2);  // attn(iq, tk, tv)

    add_ln_kernel<<<BATCH * T_SEQ, 256, 0, stream>>>(cur1, a1, ng + 0 * D_MODEL, nb + 0 * D_MODEL, x1);
    add_ln_kernel<<<BATCH * T_SEQ, 256, 0, stream>>>(cur2, a2, ng + 1 * D_MODEL, nb + 1 * D_MODEL, x2);

    ffn1_kernel<<<BATCH * 32, 256, 0, stream>>>(x1, w1i[L], i_b1, bseq, hid);
    ffn2_kernel<<<BATCH * 32, 256, 0, stream>>>(hid, w2i[L], i_b2, bseq, p1);
    ffn1_kernel<<<BATCH * 32, 256, 0, stream>>>(x2, w1t[L], t_b1, bseq, hid);
    ffn2_kernel<<<BATCH * 32, 256, 0, stream>>>(hid, w2t[L], t_b2, bseq, p2);

    float* o1 = (L == 1) ? out : h1b;
    float* o2 = (L == 1) ? out + BTD : h2b;
    add_ln_kernel<<<BATCH * T_SEQ, 256, 0, stream>>>(x1, p1, ng + 2 * D_MODEL, nb + 2 * D_MODEL, o1);
    add_ln_kernel<<<BATCH * T_SEQ, 256, 0, stream>>>(x2, p2, ng + 3 * D_MODEL, nb + 3 * D_MODEL, o2);
    cur1 = o1;
    cur2 = o2;
  }
}
